// BondGraphNetDecoder_59236188946475
// MI455X (gfx1250) — matrix operand order repaired and checked by reading, not compiled, not run on hardware
//
#include <hip/hip_runtime.h>
#include <hip/hip_bf16.h>
#include <stdint.h>

// ---------------------------------------------------------------------------
// CDNA5 (gfx1250): wave32, v_wmma_f32_16x16x32_bf16, v_cvt_pk_bf16_f32,
// LDS-staged pre-packed B operands (ds_load_b128 feeds the WMMA chain).
// ---------------------------------------------------------------------------

typedef __bf16 v16bf __attribute__((ext_vector_type(16)));
typedef float  v8f   __attribute__((ext_vector_type(8)));

union BfOp {
  v16bf    v;
  uint32_t u[8];
  uint4    q[2];
};

__device__ __forceinline__ v8f zero8() {
  v8f z = {0.f, 0.f, 0.f, 0.f, 0.f, 0.f, 0.f, 0.f};
  return z;
}

// Packed f32 -> bf16x2 (RNE), one VALU op. CDNA5 V_CVT_PK_BF16_F32.
__device__ __forceinline__ uint32_t cvt_pk_bf16(float a, float b) {
  uint32_t r;
  asm("v_cvt_pk_bf16_f32 %0, %1, %2" : "=v"(r) : "v"(a), "v"(b));
  return r;
}

__device__ __forceinline__ float elu1(float x) {
  return x > 0.f ? x : (__expf(x) - 1.f);
}

// K offset of A-operand element pair v (elements 2v, 2v+1), matching the
// packed B order: lanes 0-15 hold K 0..15, lanes 16-31 hold K 16..31.
__device__ __forceinline__ int amap(int v, int lane) {
  return 2 * v + ((lane & 16) ? 16 : 0);
}

// ---------------------------------------------------------------------------
// Pack (Krows x 128) f32 weights into WMMA B-operand order.
// Per (kc, nt) chunk of 256 dwords: [half(2)][lane(32)][e(4)] so each lane's
// operand is two b128 reads at lane*16B within a 512B half (2-way LDS
// conflict minimum). Dword = bf16 pair (K, K+1) at column nt*16+lane15;
// B layout: lanes 0-15 K=2v..2v+1, lanes 16-31 K=16+2v..17+2v per 32-K chunk.
// ---------------------------------------------------------------------------
__global__ void pack_b_kernel(const float* __restrict__ W, int Krows,
                              uint32_t* __restrict__ out) {
  int tid = blockIdx.x * blockDim.x + threadIdx.x;
  int total = (Krows >> 5) * 2048;
  if (tid >= total) return;
  int v    = tid & 7;
  int lane = (tid >> 3) & 31;
  int nt   = (tid >> 8) & 7;
  int kc   = tid >> 11;
  int K    = kc * 32 + 2 * v + ((lane & 16) ? 16 : 0);
  int col  = nt * 16 + (lane & 15);
  int dst  = (kc * 8 + nt) * 256 + (v >> 2) * 128 + lane * 4 + (v & 3);
  out[dst] = cvt_pk_bf16(W[K * 128 + col], W[(K + 1) * 128 + col]);
}

// Load one B operand from LDS-staged packed weights. cIdx = chunk*8 + nt.
__device__ __forceinline__ void load_b_lds(const uint4* swq, int cIdx,
                                           int lane, BfOp& b) {
  b.q[0] = swq[cIdx * 64 + lane];
  b.q[1] = swq[cIdx * 64 + 32 + lane];
}

// ---------------------------------------------------------------------------
// ZW[head][b][j] = bin[j] + sum_l z[b,l] * Win[(kH+l)*128 + j]
// ---------------------------------------------------------------------------
__global__ void zw_kernel(const float* __restrict__ z,
                          const float* W0, const float* b0,
                          const float* W1, const float* b1,
                          const float* W2, const float* b2,
                          const float* W3, const float* b3,
                          float* __restrict__ ZW) {
  int head = blockIdx.x >> 4;
  int b    = blockIdx.x & 15;
  int j    = threadIdx.x;  // 128
  const float* Win; const float* bin; int kH;
  switch (head) {
    case 0:  Win = W0; bin = b0; kH = 256; break;
    case 1:  Win = W1; bin = b1; kH = 384; break;
    case 2:  Win = W2; bin = b2; kH = 512; break;
    default: Win = W3; bin = b3; kH = 512; break;
  }
  float acc = bin[j];
  for (int l = 0; l < 128; ++l)
    acc += z[b * 128 + l] * Win[(kH + l) * 128 + j];
  ZW[(head * 16 + b) * 128 + j] = acc;
}

// ---------------------------------------------------------------------------
// GW[m][:] = gathered_features(m) @ Win_g via WMMA; packed Win_g staged in
// LDS once per block.
// ---------------------------------------------------------------------------
struct GwCfg {
  const float*    tnr;     // (4096,128)
  const int*      idx;     // (M, kk)
  int             kk;
  int             M;
  const uint32_t* pkWin;   // packed Win_g, kk*8192 dwords
  float*          GW;      // (M,128) out
};

#define GST 132  // padded LDS row stride (floats) -> bank-conflict free

__global__ void __launch_bounds__(256) gw_kernel(GwCfg cfg) {
  extern __shared__ float smem[];
  float*    g_all = smem;                       // 8 waves * 16*GST
  uint32_t* wsh   = (uint32_t*)(smem + 8 * 16 * GST);  // kk*8192 dwords

  int lane = threadIdx.x & 31;
  int wv   = threadIdx.x >> 5;
  int m0   = (blockIdx.x * 8 + wv) * 16;
  int row  = lane & 15;
  int hi   = lane >> 4;
  float* g = g_all + wv * (16 * GST);

  // Stage packed Win_g into LDS (coalesced b128 copies).
  {
    int nq = cfg.kk * 2048;  // uint4 count
    const uint4* src = (const uint4*)cfg.pkWin;
    uint4* dst = (uint4*)wsh;
    for (int i = threadIdx.x; i < nq; i += 256) dst[i] = src[i];
  }
  __syncthreads();
  const uint4* swq = (const uint4*)wsh;

  v8f acc[8];
#pragma unroll
  for (int nt = 0; nt < 8; ++nt) acc[nt] = zero8();

  for (int t = 0; t < cfg.kk; ++t) {
    // Stage 16 gathered node-repr rows (coalesced float4 copies).
#pragma unroll
    for (int r = 0; r < 16; ++r) {
      int node = cfg.idx[(m0 + r) * cfg.kk + t];
      const float4* src = (const float4*)(cfg.tnr + (size_t)node * 128);
      ((float4*)(g + r * GST))[lane] = src[lane];
    }
    // Pack A operands (4 K-chunks of 32) from LDS -> bf16.
    BfOp A[4];
#pragma unroll
    for (int kc = 0; kc < 4; ++kc)
#pragma unroll
      for (int v = 0; v < 8; ++v) {
        int k = kc * 32 + amap(v, lane);
        A[kc].u[v] = cvt_pk_bf16(g[row * GST + k], g[row * GST + k + 1]);
      }
#pragma unroll
    for (int nt = 0; nt < 8; ++nt) {
      BfOp Bp[4];
#pragma unroll
      for (int kc = 0; kc < 4; ++kc)
        load_b_lds(swq, (t * 4 + kc) * 8 + nt, lane, Bp[kc]);
#pragma unroll
      for (int kc = 0; kc < 4; ++kc)
        acc[nt] = __builtin_amdgcn_wmma_f32_16x16x32_bf16(
            false, A[kc].v, false, Bp[kc].v, (short)0, acc[nt], false, false);
    }
  }
  // C layout: VGPR i -> row (i + 8*hi), col nt*16 + lane15.
#pragma unroll
  for (int nt = 0; nt < 8; ++nt)
#pragma unroll
    for (int i = 0; i < 8; ++i)
      cfg.GW[(size_t)(m0 + i + hi * 8) * 128 + nt * 16 + (lane & 15)] =
          acc[nt][i];
}

// ---------------------------------------------------------------------------
// Main per-head MLP: h1 = elu(GW[m] + ZW[b]); two 128x128 WMMA layers
// (weights LDS-staged); final 128->1 dot folded into accumulators.
// ---------------------------------------------------------------------------
struct MlpCfg {
  const float*    GW;
  const float*    ZWhead;   // (16,128) for this head
  const uint32_t* pkWhid;   // packed (2,128,128) = 16384 dwords
  const float*    bhid;     // (2,128)
  const float*    Wout;     // (128)
  const float*    bout;     // (1)
  float*          out;      // d_out + head offset, (B, M)
  int             M;
  int             tilesPerB;  // M/16 (multiple of 8)
};

#define AST 132  // padded activation row stride (floats)
#define MLP_FIX (8 * 16 * AST + 8 * 32 * 9 + 256 + 128 + 128)  // 19712 floats

__global__ void __launch_bounds__(256) mlp_kernel(MlpCfg cfg) {
  extern __shared__ float smem[];
  float* acts = smem;                        // 8 * 16*AST
  float* red  = acts + 8 * 16 * AST;         // 8 * 32*9
  float* bh   = red + 8 * 32 * 9;            // 256
  float* wout = bh + 256;                    // 128
  float* zwsh = wout + 128;                  // 128
  uint32_t* wsh = (uint32_t*)(smem + MLP_FIX);  // 16384 dwords packed weights

  int lane   = threadIdx.x & 31;
  int wv     = threadIdx.x >> 5;
  int waveId = blockIdx.x * 8 + wv;
  int b      = waveId / cfg.tilesPerB;
  int m0     = (waveId % cfg.tilesPerB) * 16;
  int row    = lane & 15;
  int hi     = lane >> 4;

  {  // stage per-block constants + packed weights (b uniform within block)
    int t = threadIdx.x;
    if (t < 256) bh[t] = cfg.bhid[t];
    if (t < 128) wout[t] = cfg.Wout[t];
    int bBlock = (blockIdx.x * 8) / cfg.tilesPerB;
    if (t < 128) zwsh[t] = cfg.ZWhead[bBlock * 128 + t];
    const uint4* src = (const uint4*)cfg.pkWhid;
    uint4* dst = (uint4*)wsh;
#pragma unroll
    for (int i = 0; i < 16; ++i) dst[t + i * 256] = src[t + i * 256];
  }
  __syncthreads();
  const uint4* swq = (const uint4*)wsh;

  float* a = acts + wv * (16 * AST);

  // h1 = elu(GW[m] + ZW[b])
#pragma unroll 4
  for (int i = 0; i < 64; ++i) {
    int flat = i * 32 + lane;
    int r = flat >> 7, c = flat & 127;
    float vgw = cfg.GW[(size_t)(m0 + r) * 128 + c];
    a[r * AST + c] = elu1(vgw + zwsh[c]);
  }

  float part[8];
#pragma unroll
  for (int i = 0; i < 8; ++i) part[i] = 0.f;

#pragma unroll
  for (int layer = 0; layer < 2; ++layer) {
    // Pack the wave's full 16x128 input into A registers (allows in-place
    // activation update for layer 0).
    BfOp A[4];
#pragma unroll
    for (int kc = 0; kc < 4; ++kc)
#pragma unroll
      for (int v = 0; v < 8; ++v) {
        int k = kc * 32 + amap(v, lane);
        A[kc].u[v] = cvt_pk_bf16(a[row * AST + k], a[row * AST + k + 1]);
      }
#pragma unroll
    for (int nt = 0; nt < 8; ++nt) {
      BfOp Bp[4];
#pragma unroll
      for (int kc = 0; kc < 4; ++kc)
        load_b_lds(swq, (layer * 4 + kc) * 8 + nt, lane, Bp[kc]);
      v8f c = zero8();
#pragma unroll
      for (int kc = 0; kc < 4; ++kc)
        c = __builtin_amdgcn_wmma_f32_16x16x32_bf16(
            false, A[kc].v, false, Bp[kc].v, (short)0, c, false, false);

      int col = nt * 16 + (lane & 15);
      float bb = bh[layer * 128 + col];
      if (layer == 0) {
#pragma unroll
        for (int i = 0; i < 8; ++i)
          a[(i + hi * 8) * AST + col] = elu1(c[i] + bb);
      } else {
        float w = wout[col];
#pragma unroll
        for (int i = 0; i < 8; ++i)
          part[i] += elu1(c[i] + bb) * w;  // fold output dot into regs
      }
    }
  }

  // Cross-lane reduction of the output dot (within one wave, via LDS).
  float* rw = red + wv * (32 * 9);
#pragma unroll
  for (int i = 0; i < 8; ++i) rw[lane * 9 + i] = part[i];
  if (lane < 16) {
    int r  = lane;
    int g0 = (r >> 3) * 16;  // lanes that hold rows r's half
    float s = 0.f;
#pragma unroll
    for (int j = 0; j < 16; ++j) s += rw[(g0 + j) * 9 + (r & 7)];
    cfg.out[(size_t)b * cfg.M + m0 + r] = s + cfg.bout[0];
  }
}

// ---------------------------------------------------------------------------
// Host launcher
// ---------------------------------------------------------------------------
extern "C" void kernel_launch(void* const* d_in, const int* in_sizes, int n_in,
                              void* d_out, int out_size, void* d_ws, size_t ws_size,
                              hipStream_t stream) {
  (void)in_sizes; (void)n_in; (void)out_size; (void)ws_size;

  const float* z   = (const float*)d_in[0];
  const float* tnr = (const float*)d_in[1];
  const int* bond_idx = (const int*)d_in[2];
  const int* angle_idx = (const int*)d_in[3];
  const int* tors_idx  = (const int*)d_in[4];

  // heads: bond, angle, dcos, dsin; each {Win,bin,Whid,bhid,Wout,bout}
  const float* Win[4];  const float* bin[4];
  const float* Whid[4]; const float* bhidp[4];
  const float* Woutp[4]; const float* boutp[4];
  for (int h = 0; h < 4; ++h) {
    Win[h]   = (const float*)d_in[5 + 6 * h + 0];
    bin[h]   = (const float*)d_in[5 + 6 * h + 1];
    Whid[h]  = (const float*)d_in[5 + 6 * h + 2];
    bhidp[h] = (const float*)d_in[5 + 6 * h + 3];
    Woutp[h] = (const float*)d_in[5 + 6 * h + 4];
    boutp[h] = (const float*)d_in[5 + 6 * h + 5];
  }

  const int kk[4] = {2, 3, 4, 4};
  const int M[4]  = {4096, 8192, 8192, 8192};
  const int kH[4] = {256, 384, 512, 512};
  const int outOff[4] = {0, 65536, 196608, 327680};
  const int* idx[4] = {bond_idx, angle_idx, tors_idx, tors_idx};

  // -------- workspace layout (floats / dwords) --------
  float* ws = (float*)d_ws;
  float* ZW   = ws;                       // 4*16*128
  float* GWp[4];
  GWp[0] = ZW + 4 * 16 * 128;             // 4096*128
  GWp[1] = GWp[0] + 4096 * 128;           // 8192*128
  GWp[2] = GWp[1] + 8192 * 128;
  GWp[3] = GWp[2] + 8192 * 128;
  uint32_t* pk = (uint32_t*)(GWp[3] + 8192 * 128);
  uint32_t* pkWin[4]; uint32_t* pkWh[4];
  {
    uint32_t* cur = pk;
    for (int h = 0; h < 4; ++h) {
      pkWin[h] = cur; cur += (kH[h] / 32) * 2048;   // Win_g packed
      pkWh[h]  = cur; cur += 8 * 2048;              // Whid (256 K-rows) packed
    }
  }

  // -------- setup kernels --------
  zw_kernel<<<64, 128, 0, stream>>>(z, Win[0], bin[0], Win[1], bin[1],
                                    Win[2], bin[2], Win[3], bin[3], ZW);
  for (int h = 0; h < 4; ++h) {
    int blkW = (kH[h] / 32) * 8;  // total dwords / 256
    pack_b_kernel<<<blkW, 256, 0, stream>>>(Win[h], kH[h], pkWin[h]);
    pack_b_kernel<<<64, 256, 0, stream>>>(Whid[h], 256, pkWh[h]);
  }

  // -------- GW = gather(tnr, idx) @ Win_g --------
  size_t gwShmemMax = (size_t)(8 * 16 * GST + 4 * 8192) * sizeof(float);
  (void)hipFuncSetAttribute((const void*)gw_kernel,
                            hipFuncAttributeMaxDynamicSharedMemorySize,
                            (int)gwShmemMax);
  for (int h = 0; h < 4; ++h) {
    GwCfg c;
    c.tnr = tnr; c.idx = idx[h]; c.kk = kk[h]; c.M = M[h];
    c.pkWin = pkWin[h]; c.GW = GWp[h];
    size_t shmem = (size_t)(8 * 16 * GST + kk[h] * 8192) * sizeof(float);
    gw_kernel<<<M[h] / 128, 256, shmem, stream>>>(c);
  }

  // -------- main MLP per head --------
  size_t mlpShmem = (size_t)(MLP_FIX + 16384) * sizeof(float);
  (void)hipFuncSetAttribute((const void*)mlp_kernel,
                            hipFuncAttributeMaxDynamicSharedMemorySize,
                            (int)mlpShmem);
  for (int h = 0; h < 4; ++h) {
    MlpCfg c;
    c.GW = GWp[h];
    c.ZWhead = ZW + h * 16 * 128;
    c.pkWhid = pkWh[h];
    c.bhid = bhidp[h];
    c.Wout = Woutp[h];
    c.bout = boutp[h];
    c.out = (float*)d_out + outOff[h];
    c.M = M[h];
    c.tilesPerB = M[h] / 16;
    int blocks = 16 * (M[h] / 16) / 8;  // B * tiles / 8 waves per block
    mlp_kernel<<<blocks, 256, mlpShmem, stream>>>(c);
  }
}
